// BlockModel_82678120448388
// MI455X (gfx1250) — matrix operand order repaired and checked by reading, not compiled, not run on hardware
//
#include <hip/hip_runtime.h>
#include <hip/hip_bf16.h>
#include <cstdint>
#include <cstddef>

// Problem dims (fixed by the reference)
#define BSZ   4
#define SEQ   2048
#define EMB   512
#define BD    8
#define NH    64            // EMB/BD
#define HID   4096          // EMB*BD
#define MTOK  8192          // BSZ*SEQ
#define CHUNK 64
#define NCHK  32            // SEQ/CHUNK

typedef __attribute__((ext_vector_type(16))) __bf16 bf16x16;
typedef __attribute__((ext_vector_type(8)))  __bf16 bf16x8;
typedef __attribute__((ext_vector_type(8)))  float  v8f;

// ---------------------------------------------------------------- helpers
__device__ __forceinline__ unsigned short f2bf(float f) {
  unsigned u = __float_as_uint(f);
  u += 0x7fffu + ((u >> 16) & 1u);        // round-to-nearest-even
  return (unsigned short)(u >> 16);
}

// ---------------------------------------------------------------- f32 -> bf16
__global__ __launch_bounds__(256) void cvt_bf16(const float* __restrict__ in,
                                                unsigned short* __restrict__ out,
                                                int n) {
  int i = blockIdx.x * 256 + threadIdx.x;
  int stride = gridDim.x * 256;
  for (; i < n; i += stride) out[i] = f2bf(in[i]);
}

// ---------------------------------------------------------------- WMMA NT GEMM
// C[M,N] = A[M,K] * B[N,K]^T + bias ; optional ReLU ; f32 or bf16 output.
// Block: 256 thr = 8 waves. Block tile 64(M) x 256(N), double-buffered in LDS.
// Wave grid 2(M) x 4(N); wave tile 32 x 64 = 2 A-frags x 4 B-frags, 8 WMMA/Kstep.
// K step 32 (v_wmma_f32_16x16x32_bf16). Requires M%64==0, N%256==0, K%32==0.
template<bool RELU, bool OUT_BF16>
__global__ __launch_bounds__(256) void gemm_nt_wmma(
    const unsigned short* __restrict__ A,   // bf16, M x K row-major
    const unsigned short* __restrict__ B,   // bf16, N x K row-major
    const float* __restrict__ bias,         // N (may be null)
    void* __restrict__ Cout,
    int M, int N, int K)
{
  constexpr int LSTR = 40;                  // 32 K-halves + 8 pad (bank-tiling)
  constexpr int A_HALVES = 64 * LSTR;       // 2560
  constexpr int B_HALVES = 256 * LSTR;      // 10240
  __shared__ unsigned short smem[2 * (A_HALVES + B_HALVES)];   // 51,200 B

  const int tid  = threadIdx.x;
  const int lane = tid & 31;
  const int wave = tid >> 5;
  const int wm = wave & 1;                  // M direction (32 rows each)
  const int wn = wave >> 1;                 // N direction (64 cols each)

  const int blkM = blockIdx.y * 64;
  const int blkN = blockIdx.x * 256;

  // cooperative staging: thread -> (row = tid/4, 16B K-chunk = tid%4)
  const int trow = tid >> 2;                // 0..63
  const int tkc  = (tid & 3) * 8;           // halves: 0/8/16/24
  const unsigned short* Ag = A + (size_t)(blkM + trow) * K + tkc;
  const unsigned short* Bg = B + (size_t)(blkN + trow) * K + tkc;
  const size_t BSTEP = (size_t)64 * K;      // 64 B-rows per staging chunk

  // fragment addressing (ISA 7.12.2):
  // A 16x32: lane&15 = M row, lane>>4 selects K halves {0..7,16..23}/{8..15,24..31}
  // B 32x16: lane&15 = N col, lane>>4 selects K halves {0..7,16..23}/{8..15,24..31}
  const int arow0 = wm * 32 + (lane & 15);
  const int akb   = (lane >> 4) << 3;       // 0 or 8
  const int bcol0 = wn * 64 + (lane & 15);
  const int bkb   = (lane >> 4) << 3;       // 0 or 8

  v8f acc[2][4] = {};
  const int nsteps = K / 32;

  uint4 ar, br0, br1, br2, br3;
  // prologue: stage tile 0 into buffer 0
  ar  = *(const uint4*)(Ag);
  br0 = *(const uint4*)(Bg);
  br1 = *(const uint4*)(Bg + BSTEP);
  br2 = *(const uint4*)(Bg + 2 * BSTEP);
  br3 = *(const uint4*)(Bg + 3 * BSTEP);
  {
    unsigned short* la = smem;
    unsigned short* lb = smem + A_HALVES;
    *(uint4*)(la + trow * LSTR + tkc) = ar;
    *(uint4*)(lb + (trow      ) * LSTR + tkc) = br0;
    *(uint4*)(lb + (trow +  64) * LSTR + tkc) = br1;
    *(uint4*)(lb + (trow + 128) * LSTR + tkc) = br2;
    *(uint4*)(lb + (trow + 192) * LSTR + tkc) = br3;
  }
  __syncthreads();

  struct HX { bf16x8 lo, hi; };

  for (int s = 0; s < nsteps; ++s) {
    const int cur = s & 1;
    const unsigned short* la = smem + cur * (A_HALVES + B_HALVES);
    const unsigned short* lb = la + A_HALVES;

    // overlap: fetch next global tile into registers while doing MMAs
    if (s + 1 < nsteps) {
      const int kk = (s + 1) * 32;
      ar  = *(const uint4*)(Ag + kk);
      br0 = *(const uint4*)(Bg + kk);
      br1 = *(const uint4*)(Bg + kk + BSTEP);
      br2 = *(const uint4*)(Bg + kk + 2 * BSTEP);
      br3 = *(const uint4*)(Bg + kk + 3 * BSTEP);
    }

    bf16x16 afrag[2];
    #pragma unroll
    for (int f = 0; f < 2; ++f) {
      const unsigned short* ap = la + (arow0 + f * 16) * LSTR + akb;
      HX h;
      h.lo = *(const bf16x8*)(ap);          // K = akb .. akb+7
      h.hi = *(const bf16x8*)(ap + 16);     // K = akb+16 .. akb+23
      afrag[f] = __builtin_bit_cast(bf16x16, h);
    }
    #pragma unroll
    for (int t = 0; t < 4; ++t) {
      const unsigned short* bp = lb + (bcol0 + t * 16) * LSTR + bkb;
      HX h;
      h.lo = *(const bf16x8*)(bp);          // K = bkb .. bkb+7
      h.hi = *(const bf16x8*)(bp + 16);     // K = bkb+16 .. bkb+23
      bf16x16 bfrag = __builtin_bit_cast(bf16x16, h);
      acc[0][t] = __builtin_amdgcn_wmma_f32_16x16x32_bf16(
          false, afrag[0], false, bfrag, (short)0, acc[0][t], false, false);
      acc[1][t] = __builtin_amdgcn_wmma_f32_16x16x32_bf16(
          false, afrag[1], false, bfrag, (short)0, acc[1][t], false, false);
    }

    if (s + 1 < nsteps) {
      __syncthreads();                      // all reads of other buffer done
      unsigned short* wa = smem + (cur ^ 1) * (A_HALVES + B_HALVES);
      unsigned short* wb = wa + A_HALVES;
      *(uint4*)(wa + trow * LSTR + tkc) = ar;
      *(uint4*)(wb + (trow      ) * LSTR + tkc) = br0;
      *(uint4*)(wb + (trow +  64) * LSTR + tkc) = br1;
      *(uint4*)(wb + (trow + 128) * LSTR + tkc) = br2;
      *(uint4*)(wb + (trow + 192) * LSTR + tkc) = br3;
      __syncthreads();                      // writes visible before next reads
    }
  }

  // C/D layout: VGPR r -> row (base + r + 8*(lane>=16)), col = lane&15
  const int mlocal = (lane >> 4) << 3;
  #pragma unroll
  for (int f = 0; f < 2; ++f) {
    const int mbase = blkM + wm * 32 + f * 16 + mlocal;
    #pragma unroll
    for (int t = 0; t < 4; ++t) {
      const int n = blkN + wn * 64 + t * 16 + (lane & 15);
      const float bb = bias ? bias[n] : 0.0f;
      #pragma unroll
      for (int r = 0; r < 8; ++r) {
        float vv = acc[f][t][r] + bb;
        if (RELU) vv = fmaxf(vv, 0.0f);
        size_t o = (size_t)(mbase + r) * N + n;
        if (OUT_BF16) ((unsigned short*)Cout)[o] = f2bf(vv);
        else          ((float*)Cout)[o] = vv;
      }
    }
  }
}

// ---------------------------------------------------------------- 8x8 block norm
// blk is (MTOK, HID) f32 viewed as (tok, head, 8, 8). Per (tok,head):
//  m -= colmean ; n_j = (sum_i |m_ij|^1.2)^(1/1.2) ; m /= max_j n_j
__global__ __launch_bounds__(256) void blocknorm(float* __restrict__ blk) {
  int idx = blockIdx.x * 256 + threadIdx.x;       // tok*NH + head
  if (idx >= MTOK * NH) return;
  float* p = blk + (size_t)(idx >> 6) * HID + (idx & 63) * 64;
  float m[64];
  #pragma unroll
  for (int q = 0; q < 16; ++q) ((float4*)m)[q] = ((const float4*)p)[q];
  #pragma unroll
  for (int j = 0; j < 8; ++j) {
    float s = 0.f;
    #pragma unroll
    for (int i = 0; i < 8; ++i) s += m[i * 8 + j];
    s *= 0.125f;
    #pragma unroll
    for (int i = 0; i < 8; ++i) m[i * 8 + j] -= s;
  }
  float mx = 0.f;
  #pragma unroll
  for (int j = 0; j < 8; ++j) {
    float s = 0.f;
    #pragma unroll
    for (int i = 0; i < 8; ++i) s += __powf(fabsf(m[i * 8 + j]), 1.2f);
    mx = fmaxf(mx, __powf(s, 1.0f / 1.2f));
  }
  float inv = 1.0f / mx;
  #pragma unroll
  for (int k = 0; k < 64; ++k) m[k] *= inv;
  #pragma unroll
  for (int q = 0; q < 16; ++q) ((float4*)p)[q] = ((const float4*)m)[q];
}

// ---------------------------------------------------------------- chunked scan
// Recurrence per (b,h): state_s = A_s*state_{s-1} + v_s , state_{-1}=a0.
// unit id = ((b*NH+h)*NCHK + c). 8 threads per unit, wave32 shuffles (width 8).

// Stage A: per-chunk summary. thread j holds column j of cumulative P and u[j].
__global__ __launch_bounds__(256) void scan_stageA(
    const float* __restrict__ Ablk, const float* __restrict__ vbuf,
    float* __restrict__ Psum,       // [units][8col][8row]  (column-major 8x8)
    float* __restrict__ Usum)       // [units][8]
{
  const int unit = blockIdx.x * 32 + (threadIdx.x >> 3);
  const int j = threadIdx.x & 7;
  if (unit >= BSZ * NH * NCHK) return;
  const int c  = unit & 31;
  const int hd = (unit >> 5) & 63;
  const int b  = unit >> 11;

  float p[8];
  #pragma unroll
  for (int k = 0; k < 8; ++k) p[k] = (k == j) ? 1.0f : 0.0f;   // P = I (col j)
  float u = 0.0f;                                              // u[j]

  for (int t = 0; t < CHUNK; ++t) {
    const int s = c * CHUNK + t;
    const size_t abase = (size_t)(b * SEQ + s) * HID + hd * 64 + j * 8;
    float4 a01 = *(const float4*)(Ablk + abase);
    float4 a23 = *(const float4*)(Ablk + abase + 4);
    float a[8] = {a01.x, a01.y, a01.z, a01.w, a23.x, a23.y, a23.z, a23.w}; // A[j][k]
    // u[j] = sum_k A[j][k] * u_old[k] + v[j]
    float nu = vbuf[(size_t)(b * SEQ + s) * EMB + hd * 8 + j];
    #pragma unroll
    for (int k = 0; k < 8; ++k) nu += a[k] * __shfl(u, k, 8);
    // P[:,j] = A * P_old[:,j] ; A[i][k] via shuffle from lane i
    float np[8];
    #pragma unroll
    for (int i = 0; i < 8; ++i) {
      float acc = 0.f;
      #pragma unroll
      for (int k = 0; k < 8; ++k) acc += __shfl(a[k], i, 8) * p[k];
      np[i] = acc;
    }
    #pragma unroll
    for (int k = 0; k < 8; ++k) p[k] = np[k];
    u = nu;
  }
  #pragma unroll
  for (int i = 0; i < 8; ++i) Psum[(size_t)unit * 64 + j * 8 + i] = p[i];
  Usum[(size_t)unit * 8 + j] = u;
}

// Stage B: sequential over the 32 chunk summaries of each (b,h).
// thread i holds carry[i]. Writes Init[(b,h,c)] = state entering chunk c.
__global__ __launch_bounds__(256) void scan_stageB(
    const float* __restrict__ Psum, const float* __restrict__ Usum,
    const float* __restrict__ a0,   // (NH, BD)
    float* __restrict__ Init)       // [units][8]
{
  const int bh = blockIdx.x * 32 + (threadIdx.x >> 3);
  const int i = threadIdx.x & 7;
  if (bh >= BSZ * NH) return;
  const int hd = bh & 63;

  float carry = a0[hd * 8 + i];
  for (int c = 0; c < NCHK; ++c) {
    const size_t uu = (size_t)bh * NCHK + c;
    Init[uu * 8 + i] = carry;
    float nc = Usum[uu * 8 + i];
    #pragma unroll
    for (int j = 0; j < 8; ++j)
      nc += Psum[uu * 64 + j * 8 + i] * __shfl(carry, j, 8);
    carry = nc;
  }
}

// Stage C: replay each chunk from its corrected initial state, emit outputs.
__global__ __launch_bounds__(256) void scan_stageC(
    const float* __restrict__ Ablk, const float* __restrict__ vbuf,
    const float* __restrict__ Init, float* __restrict__ out)
{
  const int unit = blockIdx.x * 32 + (threadIdx.x >> 3);
  const int i = threadIdx.x & 7;
  if (unit >= BSZ * NH * NCHK) return;
  const int c  = unit & 31;
  const int hd = (unit >> 5) & 63;
  const int b  = unit >> 11;

  float st = Init[(size_t)unit * 8 + i];
  for (int t = 0; t < CHUNK; ++t) {
    const int s = c * CHUNK + t;
    const size_t abase = (size_t)(b * SEQ + s) * HID + hd * 64 + i * 8;
    float4 a01 = *(const float4*)(Ablk + abase);
    float4 a23 = *(const float4*)(Ablk + abase + 4);
    float a[8] = {a01.x, a01.y, a01.z, a01.w, a23.x, a23.y, a23.z, a23.w}; // A[i][k]
    float ns = vbuf[(size_t)(b * SEQ + s) * EMB + hd * 8 + i];
    #pragma unroll
    for (int k = 0; k < 8; ++k) ns += a[k] * __shfl(st, k, 8);
    st = ns;
    out[(size_t)(b * SEQ + s) * EMB + hd * 8 + i] = st;
  }
}

// ---------------------------------------------------------------- launcher
extern "C" void kernel_launch(void* const* d_in, const int* in_sizes, int n_in,
                              void* d_out, int out_size, void* d_ws, size_t ws_size,
                              hipStream_t stream)
{
  const float* x  = (const float*)d_in[0];
  const float* W1 = (const float*)d_in[1];
  const float* b1 = (const float*)d_in[2];
  const float* W2 = (const float*)d_in[3];
  const float* b2 = (const float*)d_in[4];
  const float* V1 = (const float*)d_in[5];
  const float* c1 = (const float*)d_in[6];
  const float* V2 = (const float*)d_in[7];
  const float* c2 = (const float*)d_in[8];
  const float* a0 = (const float*)d_in[9];
  float* out = (float*)d_out;

  uint8_t* base = (uint8_t*)d_ws;
  size_t off = 0;
  auto take = [&](size_t bytes) -> uint8_t* {
    off = (off + 255) & ~(size_t)255;
    uint8_t* p = base + off;
    off += bytes;
    return p;
  };

  const size_t nx = (size_t)MTOK * EMB;      // 4,194,304
  const size_t nw1 = (size_t)HID * EMB;      // 2,097,152
  const size_t nw2 = (size_t)HID * HID;      // 16,777,216
  const size_t nv  = (size_t)EMB * EMB;      // 262,144
  const size_t nh1 = (size_t)MTOK * HID;     // 33,554,432
  const size_t nhv = (size_t)MTOK * EMB;
  const size_t units = (size_t)BSZ * NH * NCHK; // 8192

  unsigned short* xb  = (unsigned short*)take(nx  * 2);
  unsigned short* w1b = (unsigned short*)take(nw1 * 2);
  unsigned short* w2b = (unsigned short*)take(nw2 * 2);
  unsigned short* v1b = (unsigned short*)take(nv  * 2);
  unsigned short* v2b = (unsigned short*)take(nv  * 2);
  unsigned short* h1b = (unsigned short*)take(nh1 * 2);
  unsigned short* hvb = (unsigned short*)take(nhv * 2);
  float* blk  = (float*)take(nh1 * 4);
  float* vbuf = (float*)take(nhv * 4);
  float* Psum = (float*)take(units * 64 * 4);
  float* Usum = (float*)take(units * 8 * 4);
  float* Init = (float*)take(units * 8 * 4);
  // total ~276 MB of d_ws used

  auto cvt = [&](const float* src, unsigned short* dst, size_t n) {
    int blocks = (int)((n + 255) / 256);
    cvt_bf16<<<blocks, 256, 0, stream>>>(src, dst, (int)n);
  };
  cvt(x,  xb,  nx);
  cvt(W1, w1b, nw1);
  cvt(W2, w2b, nw2);
  cvt(V1, v1b, nv);
  cvt(V2, v2b, nv);

  // h1 = relu(x @ W1^T + b1)           (8192 x 4096, K=512) -> bf16
  gemm_nt_wmma<true,  true ><<<dim3(HID / 256, MTOK / 64), 256, 0, stream>>>(
      xb, w1b, b1, h1b, MTOK, HID, EMB);
  // blk = h1 @ W2^T + b2               (8192 x 4096, K=4096) -> f32
  gemm_nt_wmma<false, false><<<dim3(HID / 256, MTOK / 64), 256, 0, stream>>>(
      h1b, w2b, b2, blk, MTOK, HID, HID);
  // per 8x8 block center + L1.2-norm scale
  blocknorm<<<(MTOK * NH + 255) / 256, 256, 0, stream>>>(blk);
  // hv = relu(x @ V1^T + c1)           (8192 x 512, K=512) -> bf16
  gemm_nt_wmma<true,  true ><<<dim3(EMB / 256, MTOK / 64), 256, 0, stream>>>(
      xb, v1b, c1, hvb, MTOK, EMB, EMB);
  // v = hv @ V2^T + c2                 (8192 x 512, K=512) -> f32
  gemm_nt_wmma<false, false><<<dim3(EMB / 256, MTOK / 64), 256, 0, stream>>>(
      hvb, v2b, c2, vbuf, MTOK, EMB, EMB);

  // chunked linear scan
  scan_stageA<<<(int)(units / 32), 256, 0, stream>>>(blk, vbuf, Psum, Usum);
  scan_stageB<<<(BSZ * NH + 31) / 32, 256, 0, stream>>>(Psum, Usum, a0, Init);
  scan_stageC<<<(int)(units / 32), 256, 0, stream>>>(blk, vbuf, Init, out);
}
